// ChatGLM2Attention_89223650607766
// MI455X (gfx1250) — matrix operand order repaired and checked by reading, not compiled, not run on hardware
//
#include <hip/hip_runtime.h>
#include <math.h>
#include <stdint.h>

typedef __bf16 bf16;
typedef __attribute__((ext_vector_type(8)))  __bf16 v8bf;
typedef __attribute__((ext_vector_type(16))) __bf16 v16bf;
typedef __attribute__((ext_vector_type(8)))  float  v8f;

// Problem constants
#define BATCH   2
#define SEQ     1024
#define NSTATE  4096
#define NHEAD   32
#define DHEAD   128
#define NGROUP  2
#define HPERG   16
#define QKVN    4608          // DHEAD*(NHEAD + 2*NGROUP)
#define COEFF   4.0f          // LAYER_IDX+1
#define QSCALE  0.0220970869120796f   // 1/(sqrt(128)*4)

// ---- optional CDNA5 async global->LDS path (guarded; falls back cleanly) ----
#if defined(__has_builtin)
#if __has_builtin(__builtin_amdgcn_global_load_async_to_lds_b128)
#define HAVE_ASYNC_LDS 1
#endif
#endif

#ifdef HAVE_ASYNC_LDS
// Builtin signature (from hipcc diagnostic): first param is
// 'int __attribute__((vector_size(16))) __device__ *'  i.e. vi4 addrspace(1)*.
typedef __attribute__((__vector_size__(4 * sizeof(int)))) int vi4;
typedef vi4 __attribute__((address_space(1)))* gv4p;   // global (AS1)
typedef vi4 __attribute__((address_space(3)))* lv4p;   // LDS (AS3)
#endif

__device__ __forceinline__ void async_copy16(const bf16* g, bf16* l) {
#ifdef HAVE_ASYNC_LDS
  __builtin_amdgcn_global_load_async_to_lds_b128(
      (gv4p)(uintptr_t)g,   // flat==global numerically for global memory
      (lv4p)l,              // generic -> AS3 addrspacecast
      0, 0);
#else
  *(uint4*)l = *(const uint4*)g;
#endif
}

__device__ __forceinline__ void wait_stage() {
#ifdef HAVE_ASYNC_LDS
#if __has_builtin(__builtin_amdgcn_s_wait_asynccnt)
  __builtin_amdgcn_s_wait_asynccnt(0);
#else
  asm volatile("s_wait_asynccnt 0" ::: "memory");
#endif
#endif
  __syncthreads();
}

__device__ __forceinline__ v8f zero8() {
  v8f z = {0.f,0.f,0.f,0.f,0.f,0.f,0.f,0.f};
  return z;
}

// 16x32 bf16 A-fragment for one lane (contiguous K order, matches B loads):
// lanes 0-15 hold K={0..15}, lanes 16-31 hold K={16..31} of row lane&15.
__device__ __forceinline__ v16bf load_a_frag(const bf16* p, int half) {
  v8bf lo = *(const v8bf*)(p + half * 16);
  v8bf hi = *(const v8bf*)(p + half * 16 + 8);
  return __builtin_shufflevector(lo, hi, 0,1,2,3,4,5,6,7,8,9,10,11,12,13,14,15);
}

__device__ __forceinline__ v8f wmma_bf16(v16bf a, v16bf b, v8f c) {
  return __builtin_amdgcn_wmma_f32_16x16x32_bf16(false, a, false, b,
                                                 (short)0, c, false, false);
}

// ---------------------------------------------------------------------------
// fp32 -> bf16 convert, 128-bit loads / 64-bit stores (pure bandwidth)
// ---------------------------------------------------------------------------
__global__ void cvt_bf16_kernel(const float4* __restrict__ src,
                                bf16* __restrict__ dst, size_t n4) {
  size_t i = (size_t)blockIdx.x * blockDim.x + threadIdx.x;
  const size_t stride = (size_t)gridDim.x * blockDim.x;
  for (; i < n4; i += stride) {
    float4 v = src[i];
    union { bf16 h[4]; uint2 u; } t;
    t.h[0] = (bf16)v.x; t.h[1] = (bf16)v.y;
    t.h[2] = (bf16)v.z; t.h[3] = (bf16)v.w;
    *(uint2*)(dst + 4 * i) = t.u;
  }
}

// ---------------------------------------------------------------------------
// Tiled WMMA GEMM: C[M,N] (f32) = A[M,K] (bf16) * Bw[N,K]^T (+bias)
// 256 threads (8 waves), 128x128 tile, BK=64, double-buffered LDS with
// async global->LDS staging overlapped with compute.
// ---------------------------------------------------------------------------
#define BM 128
#define BN 128
#define BK 64

__global__ __launch_bounds__(256) void gemm_bf16_kernel(
    const bf16* __restrict__ A, const bf16* __restrict__ Bw,
    const float* __restrict__ bias, float* __restrict__ C,
    int M, int N, int K)
{
  __shared__ __align__(64) bf16 As[2][BM * BK];
  __shared__ __align__(64) bf16 Bs[2][BN * BK];

  const int tid  = threadIdx.x;
  const int wave = tid >> 5;
  const int lane = tid & 31;
  const int half = lane >> 4;
  const int l16  = lane & 15;
  const int m0 = blockIdx.y * BM;
  const int n0 = blockIdx.x * BN;

  // Staging map: 2 threads per tile row, 64 B (4 x b128) each.
  const int lrow = tid >> 1;
  const int lcol = (tid & 1) * 32;                 // bf16 elements
  const bf16* Ag = A  + (size_t)(m0 + lrow) * K + lcol;
  const bf16* Bg = Bw + (size_t)(n0 + lrow) * K + lcol;
  const int soff = lrow * BK + lcol;               // LDS element offset

  v8f acc[8];
#pragma unroll
  for (int j = 0; j < 8; ++j) acc[j] = zero8();

  const int arow = wave * 16 + l16;                // this lane's A row in tile

  // Prologue: stage K-tile 0 into buffer 0
#pragma unroll
  for (int i = 0; i < 4; ++i) {
    async_copy16(Ag + i * 8, &As[0][soff] + i * 8);
    async_copy16(Bg + i * 8, &Bs[0][soff] + i * 8);
  }
  wait_stage();

  int buf = 0;
  for (int k0 = 0; k0 < K; k0 += BK) {
    const int nbuf = buf ^ 1;
    if (k0 + BK < K) {                             // stage next tile (async)
#pragma unroll
      for (int i = 0; i < 4; ++i) {
        async_copy16(Ag + k0 + BK + i * 8, &As[nbuf][soff] + i * 8);
        async_copy16(Bg + k0 + BK + i * 8, &Bs[nbuf][soff] + i * 8);
      }
      if (k0 + 2 * BK < K) {
        __builtin_prefetch((const void*)(Ag + k0 + 2 * BK), 0, 1);
        __builtin_prefetch((const void*)(Bg + k0 + 2 * BK), 0, 1);
      }
    }
    // Compute on current buffer: preload fragments, then back-to-back WMMA
#pragma unroll
    for (int kc = 0; kc < 2; ++kc) {
      v16bf af = load_a_frag(&As[buf][arow * BK + kc * 32], half);
      v16bf bfr[8];
#pragma unroll
      for (int j = 0; j < 8; ++j)
        bfr[j] = *(const v16bf*)(&Bs[buf][(j * 16 + l16) * BK + kc * 32 + half * 16]);
#pragma unroll
      for (int j = 0; j < 8; ++j)
        acc[j] = wmma_bf16(af, bfr[j], acc[j]);
    }
    wait_stage();                                  // own async done + all waves
    buf = nbuf;
  }

  // C layout: VGPR r -> M = r + 8*half, N = l16
  const int mrow = m0 + wave * 16 + half * 8;
#pragma unroll
  for (int j = 0; j < 8; ++j) {
    const int col = n0 + j * 16 + l16;
    const float bv = bias ? bias[col] : 0.0f;
#pragma unroll
    for (int r = 0; r < 8; ++r)
      C[(size_t)(mrow + r) * N + col] = acc[j][r] + bv;
  }
}

// ---------------------------------------------------------------------------
// RoPE + scale + split fused QKV (f32) into:
//   qb [B*G*H][S][D] bf16  (roped, *QSCALE)      -- WMMA A-friendly
//   kb [B*G][S][D]   bf16  (roped)               -- WMMA B-friendly for QK^T
//   vt [B*G][D][S]   bf16  (transposed)          -- WMMA B-friendly for P*V
// ---------------------------------------------------------------------------
__global__ __launch_bounds__(256) void rope_split_kernel(
    const float* __restrict__ fused,
    bf16* __restrict__ qb, bf16* __restrict__ kb, bf16* __restrict__ vt)
{
  size_t idx = (size_t)blockIdx.x * blockDim.x + threadIdx.x;
  if (idx >= (size_t)BATCH * SEQ * 36 * 64) return;
  const int i = (int)(idx & 63);
  size_t t = idx >> 6;
  const int row = (int)(t % 36); t /= 36;
  const int s   = (int)(t % SEQ);
  const int b   = (int)(t / SEQ);

  // inv_freq = 10000^(-i/64) = exp(-i * ln(10000)/64)
  const float ang = (float)s * expf(-(float)i * 0.14391156516f);
  const float cs = cosf(ang), sn = sinf(ang);

  const float* src = fused + ((size_t)(b * SEQ + s)) * QKVN + row * DHEAD + 2 * i;
  const float f0 = src[0], f1 = src[1];

  if (row < 32) {                 // Q: head = row, rope + scale
    const float o0 = (f0 * cs - f1 * sn) * QSCALE;
    const float o1 = (f0 * sn + f1 * cs) * QSCALE;
    bf16* dst = qb + (((size_t)(b * 32 + row) * SEQ) + s) * DHEAD + 2 * i;
    dst[0] = (bf16)o0; dst[1] = (bf16)o1;
  } else if (row < 34) {          // K: group = row-32, rope
    const int g = row - 32;
    const float o0 = f0 * cs - f1 * sn;
    const float o1 = f0 * sn + f1 * cs;
    bf16* dst = kb + (((size_t)(b * NGROUP + g) * SEQ) + s) * DHEAD + 2 * i;
    dst[0] = (bf16)o0; dst[1] = (bf16)o1;
  } else {                        // V: group = row-34, transpose into [d][s]
    const int g = row - 34;
    bf16* dst = vt + (size_t)(b * NGROUP + g) * DHEAD * SEQ;
    dst[(size_t)(2 * i)     * SEQ + s] = (bf16)f0;
    dst[(size_t)(2 * i + 1) * SEQ + s] = (bf16)f1;
  }
}

// ---------------------------------------------------------------------------
// Fused flash attention: one wave per 16-query tile of one (b,g,h).
// Key tiles of 32 (matches WMMA K=32 for P*V). Online softmax.
// ---------------------------------------------------------------------------
__global__ __launch_bounds__(128) void attn_kernel(
    const bf16* __restrict__ qb, const bf16* __restrict__ kb,
    const bf16* __restrict__ vt, bf16* __restrict__ attnb)
{
  __shared__ __align__(64) bf16 p_lds[4][16 * 32];   // per-wave P tile

  const int lane = threadIdx.x & 31;
  const int wave = threadIdx.x >> 5;
  const int half = lane >> 4;
  const int l16  = lane & 15;

  const int bgh = blockIdx.x;            // b*32 + g*16 + h
  const int b = bgh >> 5;
  const int g = (bgh >> 4) & 1;
  const int h = bgh & 15;
  const int bg = b * NGROUP + g;
  const int qtile = blockIdx.y * 4 + wave;
  const int q0 = qtile * 16;

  const bf16* qptr = qb + (size_t)bgh * SEQ * DHEAD;
  const bf16* kptr = kb + (size_t)bg * SEQ * DHEAD;
  const bf16* vptr = vt + (size_t)bg * DHEAD * SEQ;

  // Q A-fragments: row = q0 + l16, 4 K-chunks of 32 over DHEAD=128
  v16bf qa[4];
#pragma unroll
  for (int c = 0; c < 4; ++c)
    qa[c] = load_a_frag(qptr + (size_t)(q0 + l16) * DHEAD + c * 32, half);

  v8f o_acc[8];
#pragma unroll
  for (int j = 0; j < 8; ++j) o_acc[j] = zero8();
  float m_run[8], l_run[8];
#pragma unroll
  for (int r = 0; r < 8; ++r) { m_run[r] = -1e30f; l_run[r] = 0.0f; }

  const int ntiles = (q0 + 16 + 31) >> 5;  // causal: keys < q0+16
  for (int t = 0; t < ntiles; ++t) {
    const int kbase = t * 32;

    // S = Q * K^T: preload all 8 K fragments, then back-to-back WMMA
    v16bf kfr[8];
#pragma unroll
    for (int c = 0; c < 4; ++c) {
      kfr[2*c]   = *(const v16bf*)(kptr + (size_t)(kbase + l16)      * DHEAD + c * 32 + half * 16);
      kfr[2*c+1] = *(const v16bf*)(kptr + (size_t)(kbase + 16 + l16) * DHEAD + c * 32 + half * 16);
    }
    v8f s0 = zero8(), s1 = zero8();
#pragma unroll
    for (int c = 0; c < 4; ++c) {
      s0 = wmma_bf16(qa[c], kfr[2*c],   s0);
      s1 = wmma_bf16(qa[c], kfr[2*c+1], s1);
    }

    // Online softmax (rows live within 16-lane halves of the wave)
    float alpha[8];
#pragma unroll
    for (int r = 0; r < 8; ++r) {
      const int qrow = q0 + r + half * 8;
      float a0 = s0[r] * COEFF;
      float a1 = s1[r] * COEFF;
      if (kbase + l16 > qrow)      a0 = -1e30f;   // causal mask
      if (kbase + 16 + l16 > qrow) a1 = -1e30f;
      float rx = fmaxf(a0, a1);
#pragma unroll
      for (int off = 8; off >= 1; off >>= 1)
        rx = fmaxf(rx, __shfl_xor(rx, off, 32));
      const float mnew = fmaxf(m_run[r], rx);
      alpha[r] = __expf(m_run[r] - mnew);
      m_run[r] = mnew;
      const float p0 = __expf(a0 - mnew);
      const float p1 = __expf(a1 - mnew);
      float rs = p0 + p1;
#pragma unroll
      for (int off = 8; off >= 1; off >>= 1)
        rs += __shfl_xor(rs, off, 32);
      l_run[r] = l_run[r] * alpha[r] + rs;
      // write P (bf16) into per-wave LDS tile, row-major [16][32]
      const int prow = r + half * 8;
      p_lds[wave][prow * 32 + l16]      = (bf16)p0;
      p_lds[wave][prow * 32 + 16 + l16] = (bf16)p1;
    }
#pragma unroll
    for (int j = 0; j < 8; ++j)
#pragma unroll
      for (int r = 0; r < 8; ++r)
        o_acc[j][r] *= alpha[r];

    // Wave-private LDS round trip (LDS ops in-order per wave; compiler fence)
    asm volatile("s_wait_dscnt 0" ::: "memory");

    // O += P * V: preload all 8 V fragments, then back-to-back WMMA
    v16bf pa = load_a_frag(&p_lds[wave][l16 * 32], half);
    v16bf vfr[8];
#pragma unroll
    for (int j = 0; j < 8; ++j)
      vfr[j] = *(const v16bf*)(vptr + (size_t)(j * 16 + l16) * SEQ + kbase + half * 16);
#pragma unroll
    for (int j = 0; j < 8; ++j)
      o_acc[j] = wmma_bf16(pa, vfr[j], o_acc[j]);
  }

  // Normalize and store: attnb[b*S + q][head*128 + d]
#pragma unroll
  for (int r = 0; r < 8; ++r) {
    const float inv = 1.0f / l_run[r];
    const int qrow = q0 + r + half * 8;
    bf16* dst = attnb + ((size_t)(b * SEQ + qrow)) * (NHEAD * DHEAD)
                      + (size_t)(g * HPERG + h) * DHEAD;
#pragma unroll
    for (int j = 0; j < 8; ++j)
      dst[j * 16 + l16] = (bf16)(o_acc[j][r] * inv);
  }
}

// ---------------------------------------------------------------------------
// Launch
// ---------------------------------------------------------------------------
extern "C" void kernel_launch(void* const* d_in, const int* in_sizes, int n_in,
                              void* d_out, int out_size, void* d_ws, size_t ws_size,
                              hipStream_t stream) {
  const float* x      = (const float*)d_in[0];
  // d_in[1] freqs_cis, d_in[2] attention_mask: computed analytically in-kernel
  const float* qkv_w  = (const float*)d_in[3];
  const float* qkv_b  = (const float*)d_in[4];
  const float* o_w    = (const float*)d_in[5];
  float* out = (float*)d_out;

  const size_t N_X   = (size_t)BATCH * SEQ * NSTATE;
  const size_t N_WQ  = (size_t)QKVN * NSTATE;
  const size_t N_WO  = (size_t)NSTATE * NSTATE;
  const size_t N_FU  = (size_t)BATCH * SEQ * QKVN;
  const size_t N_QB  = (size_t)BATCH * NHEAD * SEQ * DHEAD;
  const size_t N_KB  = (size_t)BATCH * NGROUP * SEQ * DHEAD;

  char* ws = (char*)d_ws;
  bf16*  xb    = (bf16*)ws;                 ws += N_X  * sizeof(bf16);
  bf16*  wqb   = (bf16*)ws;                 ws += N_WQ * sizeof(bf16);
  bf16*  wob   = (bf16*)ws;                 ws += N_WO * sizeof(bf16);
  float* fused = (float*)ws;                ws += N_FU * sizeof(float);
  bf16*  qbuf  = (bf16*)ws;                 ws += N_QB * sizeof(bf16);
  bf16*  kbuf  = (bf16*)ws;                 ws += N_KB * sizeof(bf16);
  bf16*  vtb   = (bf16*)ws;                 ws += N_KB * sizeof(bf16);
  bf16*  attnb = (bf16*)ws;

  // 1. fp32 -> bf16 operand conversion (vectorized)
  cvt_bf16_kernel<<<4096, 256, 0, stream>>>((const float4*)x,     xb,  N_X  / 4);
  cvt_bf16_kernel<<<4096, 256, 0, stream>>>((const float4*)qkv_w, wqb, N_WQ / 4);
  cvt_bf16_kernel<<<4096, 256, 0, stream>>>((const float4*)o_w,   wob, N_WO / 4);

  // 2. QKV projection: fused[2048,4608] = x * qkv_w^T + qkv_b
  gemm_bf16_kernel<<<dim3(QKVN / BN, (BATCH * SEQ) / BM), 256, 0, stream>>>(
      xb, wqb, qkv_b, fused, BATCH * SEQ, QKVN, NSTATE);

  // 3. RoPE + scale + split + V transpose
  {
    const size_t total = (size_t)BATCH * SEQ * 36 * 64;
    rope_split_kernel<<<(unsigned)((total + 255) / 256), 256, 0, stream>>>(
        fused, qbuf, kbuf, vtb);
  }

  // 4. Fused flash attention (one wave per 16-query tile)
  attn_kernel<<<dim3(BATCH * NHEAD, SEQ / 16 / 4), 128, 0, stream>>>(
      qbuf, kbuf, vtb, attnb);

  // 5. Output projection: out[2048,4096] = attn * o_w^T
  gemm_bf16_kernel<<<dim3(NSTATE / BN, (BATCH * SEQ) / BM), 256, 0, stream>>>(
      attnb, wob, nullptr, out, BATCH * SEQ, NSTATE, NSTATE);
}
